// PerPositionTemporalMemory_56762287784291
// MI455X (gfx1250) — matrix operand order repaired and checked by reading, not compiled, not run on hardware
//
#include <hip/hip_runtime.h>
#include <hip/hip_bf16.h>
#include <stdint.h>

// Problem constants (from reference setup_inputs)
#define BB   8
#define TT   4096
#define TOUT 4097
#define DD   1024
#define HH   32
#define MM   (BB * TOUT)        // 32776 valid output rows
#define MPAD (129 * 256)        // 33024: M padded to the 256-row block grid

typedef __attribute__((ext_vector_type(16))) __bf16 v16bf;
typedef __attribute__((ext_vector_type(8)))  float  v8f;
typedef unsigned short u16;
typedef __attribute__((ext_vector_type(4)))  u16   v4u;
typedef __attribute__((ext_vector_type(8)))  u16   v8u;
typedef __attribute__((ext_vector_type(16))) u16   v16u;
typedef __attribute__((ext_vector_type(4)))  float v4f;

static __device__ __forceinline__ u16 f2bf(float f) {
  uint32_t u = __builtin_bit_cast(uint32_t, f);
  uint32_t r = u + 0x7FFFu + ((u >> 16) & 1u);   // round-to-nearest-even
  return (u16)(r >> 16);
}

// ---------------------------------------------------------------- K0a: f32 -> bf16 cast
__global__ __launch_bounds__(256)
void cvt_bf16_kernel(const float* __restrict__ src, u16* __restrict__ dst, int n) {
  int i = (blockIdx.x * 256 + threadIdx.x) * 4;
  if (i + 3 < n) {
    v4f v = *(const v4f*)(src + i);
    v4u o;
    o.x = f2bf(v.x); o.y = f2bf(v.y); o.z = f2bf(v.z); o.w = f2bf(v.w);
    *(v4u*)(dst + i) = o;
  }
}

// ---------------------------------------------------------------- K0c: bias1 = bv * sum(decay_w)
__global__ __launch_bounds__(256)
void bias_scale_kernel(const float* __restrict__ bv, const float* __restrict__ w,
                       float* __restrict__ out, int n) {
  float s = 0.f;
  #pragma unroll
  for (int i = 0; i < HH; ++i) s += w[i];
  int i = blockIdx.x * 256 + threadIdx.x;
  if (i < n) out[i] = bv[i] * s;
}

// ---------------------------------------------------------------- K0d: zero the M-pad rows of s
__global__ __launch_bounds__(256)
void zero_pad_kernel(u16* __restrict__ s) {
  const size_t n = (size_t)(MPAD - MM) * DD;         // 253952 u16
  size_t idx = ((size_t)blockIdx.x * 256 + threadIdx.x) * 8;
  if (idx < n) {
    v8u z = {0, 0, 0, 0, 0, 0, 0, 0};
    *(v8u*)(s + (size_t)MM * DD + idx) = z;
  }
}

// ---------------------------------------------------------------- K0b: causal decay window -> s (bf16)
// s[b,t,d] = sum_{i=0}^{31} w[i] * x[b, t-32+i, d]  (OOB x -> 0), t in [0,4096]
__global__ __launch_bounds__(256)
void decay_conv_kernel(const float* __restrict__ x, const float* __restrict__ w,
                       u16* __restrict__ s) {
  const int d  = threadIdx.x * 4;      // 256 threads cover 1024 channels
  const int b  = blockIdx.y;
  const int t0 = blockIdx.x * 8;

  float wr[HH];
  #pragma unroll
  for (int i = 0; i < HH; ++i) wr[i] = w[i];

  v4f acc[8];
  #pragma unroll
  for (int j = 0; j < 8; ++j) acc[j] = (v4f){0.f, 0.f, 0.f, 0.f};

  const float* xb = x + (size_t)b * TT * DD + d;
  #pragma unroll
  for (int rr = 0; rr < HH + 7; ++rr) {          // x row r = t0 - 32 + rr
    const int r = t0 - HH + rr;
    v4f xv = (v4f){0.f, 0.f, 0.f, 0.f};
    if (r >= 0 && r < TT) xv = *(const v4f*)(xb + (size_t)r * DD);
    #pragma unroll
    for (int j = 0; j < 8; ++j) {
      const int i = rr - j;                      // compile-time after unroll
      if (i >= 0 && i < HH) acc[j] += wr[i] * xv;
    }
  }

  u16* sb = s + ((size_t)b * TOUT + t0) * DD + d;
  #pragma unroll
  for (int j = 0; j < 8; ++j) {
    if (t0 + j < TOUT) {
      v4u o;
      o.x = f2bf(acc[j].x); o.y = f2bf(acc[j].y);
      o.z = f2bf(acc[j].z); o.w = f2bf(acc[j].w);
      *(v4u*)(sb + (size_t)j * DD) = o;
    }
  }
}

// ---------------------------------------------------------------- GEMM: C[m,n] = sum_k A[m,k]*B[n,k] + bias[n]
// A: MPAD x 1024 bf16 row-major (pad rows valid memory). B: 1024 x 1024 bf16 (n rows, k contig).
// Block tile 256x128, BK=32, 8 waves (wave32) in 4(M) x 2(N) grid; wave tile 64x64 = 4x4 WMMA
// subtiles (16 v8f accumulators). Staging via GLOBAL_LOAD_ASYNC_TO_LDS_B128, double-buffered
// LDS, ASYNCcnt-paced depth-2 pipeline. Per K-step per wave: 16 ds_read_b128 : 16 WMMA.
#define BM 256
#define BN 128
#define BK 32
#define LDSP 40                      // padded LDS row stride (u16): 80B, conflict-free b128
#define BUFU ((BM + BN) * LDSP)      // u16 per LDS buffer = 15360
#define BUFB (BUFU * 2)              // bytes per LDS buffer = 30720
#define ABYTES (BM * LDSP * 2)       // byte offset of B region inside a buffer

template <int OUT_BF16>
__global__ __launch_bounds__(256)
void gemm_bf16(const u16* __restrict__ A, const u16* __restrict__ Bm,
               const float* __restrict__ bias, void* __restrict__ Cp, int Mstore) {
  __shared__ u16 smem[2 * BUFU];     // 61440 B

  const int tid  = threadIdx.x;
  const int lane = tid & 31;
  const int wave = tid >> 5;
  const int wm   = wave >> 1;        // 0..3  (64-row slab)
  const int wn   = wave & 1;         // 0..1  (64-col slab)
  const int l16  = lane & 15;
  const int lh   = lane >> 4;        // half-wave select

  const int m0 = blockIdx.x * BM;
  const int n0 = blockIdx.y * BN;

  v8f acc[4][4];
  #pragma unroll
  for (int mi = 0; mi < 4; ++mi)
    #pragma unroll
    for (int ni = 0; ni < 4; ++ni)
      acc[mi][ni] = (v8f){0.f,0.f,0.f,0.f,0.f,0.f,0.f,0.f};

  // Staging: thread owns 16B chunks at rows row0+{0,64,128,192} (A) and row0+{0,64} (B).
  const int row0 = tid >> 2;         // 0..63
  const int seg  = (tid & 3) * 8;    // k offset (elements)

  // LDS byte addresses (low 32 bits of a flat LDS pointer == LDS offset).
  const uint32_t ldsbase = (uint32_t)(uintptr_t)(&smem[0]);
  const uint32_t rb = (uint32_t)(row0 * LDSP + seg) * 2u;
  const uint32_t rstep = 64u * LDSP * 2u;                 // 64 rows in LDS bytes
  uint32_t aL[4], bL[2];
  #pragma unroll
  for (int j = 0; j < 4; ++j) aL[j] = ldsbase + rb + (uint32_t)j * rstep;
  #pragma unroll
  for (int j = 0; j < 2; ++j) bL[j] = ldsbase + ABYTES + rb + (uint32_t)j * rstep;

  // Global byte offsets (32-bit voffset against SGPR base), k = 0.
  uint32_t aG[4], bG[2];
  #pragma unroll
  for (int j = 0; j < 4; ++j) aG[j] = ((uint32_t)(m0 + row0 + 64 * j) * DD + seg) * 2u;
  #pragma unroll
  for (int j = 0; j < 2; ++j) bG[j] = ((uint32_t)(n0 + row0 + 64 * j) * DD + seg) * 2u;

  auto issue_tile = [&](int kk, int buf) {
    const uint32_t ko = (uint32_t)kk * (BK * 2u);          // 64 B per K-step
    const uint32_t bo = (uint32_t)buf * BUFB;
    #pragma unroll
    for (int j = 0; j < 4; ++j)
      asm volatile("global_load_async_to_lds_b128 %0, %1, %2"
                   :: "v"(aL[j] + bo), "v"(aG[j] + ko), "s"(A) : "memory");
    #pragma unroll
    for (int j = 0; j < 2; ++j)
      asm volatile("global_load_async_to_lds_b128 %0, %1, %2"
                   :: "v"(bL[j] + bo), "v"(bG[j] + ko), "s"(Bm) : "memory");
  };

  const int kiters = DD / BK;        // 32
  issue_tile(0, 0);                  // depth-2 pipeline prologue
  issue_tile(1, 1);

  for (int kk = 0; kk < kiters; ++kk) {
    // Drain this tile's 6 asyncs (next tile's 6 may stay in flight).
    if (kk < kiters - 1) asm volatile("s_wait_asynccnt 0x6" ::: "memory");
    else                 asm volatile("s_wait_asynccnt 0x0" ::: "memory");
    __syncthreads();                 // tile kk visible to all waves

    const u16* Ab = smem + (kk & 1) * BUFU;
    const u16* Bb = Ab + BM * LDSP;

    // B fragments: lanes 0-15 hold K 0..15 (col = lane), lanes 16-31 hold K 16..31.
    v16bf bfr[4];
    #pragma unroll
    for (int ni = 0; ni < 4; ++ni) {
      const int rowB = wn * 64 + ni * 16 + l16;
      const v8u lo = *(const v8u*)(&Bb[rowB * LDSP + lh * 16]);
      const v8u hi = *(const v8u*)(&Bb[rowB * LDSP + lh * 16 + 8]);
      v16u t;
      #pragma unroll
      for (int e = 0; e < 8; ++e) { t[e] = lo[e]; t[e + 8] = hi[e]; }
      bfr[ni] = __builtin_bit_cast(v16bf, t);
    }

    // A fragments: lanes 0-15 hold K 0..15; lanes 16-31 hold K 16..31.
    #pragma unroll
    for (int mi = 0; mi < 4; ++mi) {
      const int rowA = wm * 64 + mi * 16 + l16;
      const v8u lo = *(const v8u*)(&Ab[rowA * LDSP + lh * 16]);
      const v8u hi = *(const v8u*)(&Ab[rowA * LDSP + lh * 16 + 8]);
      v16u t;
      #pragma unroll
      for (int e = 0; e < 8; ++e) { t[e] = lo[e]; t[e + 8] = hi[e]; }
      const v16bf afr = __builtin_bit_cast(v16bf, t);
      #pragma unroll
      for (int ni = 0; ni < 4; ++ni) {
        acc[mi][ni] = __builtin_amdgcn_wmma_f32_16x16x32_bf16(
            false, afr, false, bfr[ni], (short)0, acc[mi][ni], false, false);
      }
    }

    __syncthreads();                 // everyone finished reading buf (kk&1)
    if (kk + 2 < kiters) issue_tile(kk + 2, kk & 1);   // refill freed buffer
  }

  // Epilogue: C layout — v8f element r: row = r + 8*lh, col = l16.
  #pragma unroll
  for (int ni = 0; ni < 4; ++ni) {
    const int n  = n0 + wn * 64 + ni * 16 + l16;
    const float bn = bias[n];
    #pragma unroll
    for (int mi = 0; mi < 4; ++mi) {
      #pragma unroll
      for (int r = 0; r < 8; ++r) {
        const int m = m0 + wm * 64 + mi * 16 + lh * 8 + r;
        if (OUT_BF16) {              // att buffer is MPAD rows: no guard needed
          ((u16*)Cp)[(size_t)m * DD + n] = f2bf(acc[mi][ni][r] + bn);
        } else if (m < Mstore) {
          ((float*)Cp)[(size_t)m * DD + n] = acc[mi][ni][r] + bn;
        }
      }
    }
  }
}

// ---------------------------------------------------------------- launch
extern "C" void kernel_launch(void* const* d_in, const int* in_sizes, int n_in,
                              void* d_out, int out_size, void* d_ws, size_t ws_size,
                              hipStream_t stream) {
  (void)in_sizes; (void)n_in; (void)out_size; (void)ws_size;
  const float* x  = (const float*)d_in[0];
  const float* Wv = (const float*)d_in[1];
  const float* bv = (const float*)d_in[2];
  const float* Wo = (const float*)d_in[3];
  const float* bo = (const float*)d_in[4];
  const float* dw = (const float*)d_in[5];

  // Workspace layout (256B aligned): Wv_bf16 | Wo_bf16 | bias1 | s_bf16 | att_bf16
  uint8_t* ws = (uint8_t*)d_ws;
  size_t off = 0;
  auto take = [&](size_t bytes) -> void* {
    void* p = ws + off;
    off += (bytes + 255) & ~(size_t)255;
    return p;
  };
  u16*   Wv_bf  = (u16*)  take((size_t)DD * DD * 2);
  u16*   Wo_bf  = (u16*)  take((size_t)DD * DD * 2);
  float* bias1  = (float*)take((size_t)DD * 4);
  u16*   s_bf   = (u16*)  take((size_t)MPAD * DD * 2);
  u16*   att_bf = (u16*)  take((size_t)MPAD * DD * 2);

  cvt_bf16_kernel<<<DD * DD / 1024, 256, 0, stream>>>(Wv, Wv_bf, DD * DD);
  cvt_bf16_kernel<<<DD * DD / 1024, 256, 0, stream>>>(Wo, Wo_bf, DD * DD);
  bias_scale_kernel<<<DD / 256, 256, 0, stream>>>(bv, dw, bias1, DD);
  zero_pad_kernel<<<((MPAD - MM) * DD / 8 + 255) / 256, 256, 0, stream>>>(s_bf);
  decay_conv_kernel<<<dim3((TOUT + 7) / 8, BB), 256, 0, stream>>>(x, dw, s_bf);

  dim3 g(MPAD / BM, DD / BN);        // 129 x 8
  gemm_bf16<1><<<g, 256, 0, stream>>>(s_bf,   Wv_bf, bias1, (void*)att_bf, MPAD);
  gemm_bf16<0><<<g, 256, 0, stream>>>(att_bf, Wo_bf, bo,    d_out,         MM);
}
